// GHMT_41747082117529
// MI455X (gfx1250) — matrix operand order repaired and checked by reading, not compiled, not run on hardware
//
#include <hip/hip_runtime.h>

// ---------------------------------------------------------------------------
// MI455X (gfx1250) implementation of the GAT + relation-memory GNN.
// Dense GEMMs: v_wmma_f32_16x16x32_bf16, one 16x64 output tile per wave32.
// A and B are pre-converted to bf16 so the GEMM inner loop is pure
// global_load_b128 -> v_wmma (no cvt chains, low register pressure).
// Irregular work (edge softmax, segment sums, layernorms) is wave32
// scalar/atomic code.
// ---------------------------------------------------------------------------

#define DINL __device__ __forceinline__

constexpr int NU  = 20000;   // users
constexpr int NI  = 20000;   // items
constexpr int NN  = 40000;   // all nodes
constexpr int DD  = 128;
constexpr int HD  = 512;     // H * D
constexpr int MM  = 8;
constexpr int TT  = 5;
constexpr int EUI = 40000;
constexpr int ET  = 16000;

typedef __attribute__((ext_vector_type(16))) __bf16 v16bf;
typedef __attribute__((ext_vector_type(8)))  __bf16 v8bf;
typedef __attribute__((ext_vector_type(8)))  float  v8f;

DINL float leakyf(float x) { return x > 0.f ? x : 0.2f * x; }
DINL float eluf(float x)   { return x > 0.f ? x : (__expf(x) - 1.f); }

// ---- f32 -> bf16 conversion (8 elements / thread) -------------------------
__global__ void k_cvt(const float* __restrict__ in, __bf16* __restrict__ out, int total8) {
  int i = blockIdx.x * blockDim.x + threadIdx.x;
  if (i >= total8) return;
  const float4* p = reinterpret_cast<const float4*>(in) + (size_t)i * 2;
  float4 a = p[0], b = p[1];
  v8bf o;
  o[0]=(__bf16)a.x; o[1]=(__bf16)a.y; o[2]=(__bf16)a.z; o[3]=(__bf16)a.w;
  o[4]=(__bf16)b.x; o[5]=(__bf16)b.y; o[6]=(__bf16)b.z; o[7]=(__bf16)b.w;
  reinterpret_cast<v8bf*>(out)[i] = o;
}

// C(Mr x Nc) = epi( A(Mr x K) @ B(Nc x K)^T ), A/B bf16, C f32.
// One 16x64 tile per wave32; A frag reused across 4 v_wmma issues.
// epilogue: +bias[col], act (0 none / 1 elu), * rowscale[row*rsStride],
// accumulate (C += v) or store (C = v).
__global__ void k_gemm(const __bf16* __restrict__ A, const __bf16* __restrict__ B,
                       float* __restrict__ C, int Mr, int Nc, int K,
                       const float* __restrict__ bias, int act,
                       const float* __restrict__ rowscale, int rsStride,
                       int accumulate) {
  int wave = (int)((blockIdx.x * blockDim.x + threadIdx.x) >> 5);
  int lane = threadIdx.x & 31;
  int tilesN = Nc >> 6;                       // 64-wide column tiles
  int tot = (Mr >> 4) * tilesN;
  if (wave >= tot) return;                    // wave-uniform: EXEC stays all-ones
  int r0 = (wave / tilesN) << 4;
  int c0 = (wave % tilesN) << 6;
  int half = lane >> 4, l16 = lane & 15;
  // A 16x32 bf16 frag: lane half h holds contiguous K run [16h, 16h+16)
  const __bf16* arow = A + (size_t)(r0 + l16) * K;
  // B 32x16 bf16 frag: lane half h holds contiguous K run [16h, 16h+16)
  const __bf16* brow = B + (size_t)(c0 + l16) * K + 16 * half;
  size_t bstep = (size_t)16 * K;              // next 16-column tile of B
  v8f acc[4] = {};
  for (int k0 = 0; k0 < K; k0 += 32) {
    v8bf alo = *reinterpret_cast<const v8bf*>(arow + k0 + 16 * half);
    v8bf ahi = *reinterpret_cast<const v8bf*>(arow + k0 + 16 * half + 8);
    v16bf af = __builtin_shufflevector(alo, ahi, 0,1,2,3,4,5,6,7,8,9,10,11,12,13,14,15);
    v16bf b0 = *reinterpret_cast<const v16bf*>(brow + k0);
    v16bf b1 = *reinterpret_cast<const v16bf*>(brow + bstep + k0);
    v16bf b2 = *reinterpret_cast<const v16bf*>(brow + 2 * bstep + k0);
    v16bf b3 = *reinterpret_cast<const v16bf*>(brow + 3 * bstep + k0);
    acc[0] = __builtin_amdgcn_wmma_f32_16x16x32_bf16(false, af, false, b0, (short)0, acc[0], false, false);
    acc[1] = __builtin_amdgcn_wmma_f32_16x16x32_bf16(false, af, false, b1, (short)0, acc[1], false, false);
    acc[2] = __builtin_amdgcn_wmma_f32_16x16x32_bf16(false, af, false, b2, (short)0, acc[2], false, false);
    acc[3] = __builtin_amdgcn_wmma_f32_16x16x32_bf16(false, af, false, b3, (short)0, acc[3], false, false);
  }
#pragma unroll
  for (int j = 0; j < 4; ++j) {
    int col = c0 + 16 * j + l16;
    float bv = bias ? bias[col] : 0.f;
#pragma unroll
    for (int r = 0; r < 8; ++r) {
      int row = r0 + r + 8 * half;            // C/D layout: VGPR r -> rows r, r+8
      float v = acc[j][r] + bv;
      if (act == 1) v = eluf(v);
      if (rowscale) v *= rowscale[(size_t)row * rsStride];
      size_t idx = (size_t)row * Nc + col;
      if (accumulate) C[idx] += v; else C[idx] = v;
    }
  }
}

// el/er: out[n*4+h] = sum_d f[n,h*128+d] * attn[h,d]   (f is (n,512))
__global__ void k_attn_score(const float* __restrict__ f,
                             const float* __restrict__ attn,
                             float* __restrict__ out, int total) {
  int i = blockIdx.x * blockDim.x + threadIdx.x;
  if (i >= total) return;
  const float* fr = f + (size_t)i * 128;   // (n,h) flattening matches layout
  const float* ar = attn + (size_t)(i & 3) * 128;
  float s = 0.f;
  for (int d = 0; d < 128; ++d) s += fr[d] * ar[d];
  out[i] = s;
}

DINL unsigned fkey(float f) {
  unsigned u = __float_as_uint(f);
  return (u & 0x80000000u) ? ~u : (u | 0x80000000u);
}
DINL float fkeydec(unsigned k) {
  unsigned u = (k & 0x80000000u) ? (k & 0x7FFFFFFFu) : ~k;
  return __uint_as_float(u);
}

__global__ void k_edge_max(const float* __restrict__ el, const float* __restrict__ er,
                           const int* __restrict__ sidx, const int* __restrict__ didx,
                           float* __restrict__ esc, unsigned* __restrict__ emk, int E) {
  int i = blockIdx.x * blockDim.x + threadIdx.x;
  if (i >= E * 4) return;
  int e = i >> 2, h = i & 3;
  float v = leakyf(el[(size_t)sidx[e] * 4 + h] + er[(size_t)didx[e] * 4 + h]);
  esc[i] = v;
  atomicMax(&emk[(size_t)didx[e] * 4 + h], fkey(v));
}

__global__ void k_edge_exp(const float* __restrict__ esc, const unsigned* __restrict__ emk,
                           const int* __restrict__ didx, float* __restrict__ exb,
                           float* __restrict__ den, int E) {
  int i = blockIdx.x * blockDim.x + threadIdx.x;
  if (i >= E * 4) return;
  int e = i >> 2, h = i & 3;
  float ex = __expf(esc[i] - fkeydec(emk[(size_t)didx[e] * 4 + h]));
  exb[i] = ex;
  atomicAdd(&den[(size_t)didx[e] * 4 + h], ex);
}

// one wave per (edge,h): rst[dst,h*128+d] += a * fs[src,h*128+d]
__global__ void k_gat_agg(const float* __restrict__ exb, const float* __restrict__ den,
                          const float* __restrict__ fs, const int* __restrict__ sidx,
                          const int* __restrict__ didx, float* __restrict__ rst, int E) {
  int w = (int)((blockIdx.x * blockDim.x + threadIdx.x) >> 5);
  int lane = threadIdx.x & 31;
  if (w >= E * 4) return;
  int e = w >> 2, h = w & 3;
  float a = exb[w] / den[(size_t)didx[e] * 4 + h];
  const float* src = fs + (size_t)sidx[e] * 512 + h * 128;
  float* dst = rst + (size_t)didx[e] * 512 + h * 128;
  for (int j = lane; j < 128; j += 32) atomicAdd(&dst[j], a * src[j]);
}

// gu(bf16) = elu(rst + bias)   (rst f32 input stays intact)
__global__ void k_bias_elu_bf(const float* __restrict__ buf, const float* __restrict__ bias,
                              __bf16* __restrict__ out, int total) {
  int i = blockIdx.x * blockDim.x + threadIdx.x;
  if (i >= total) return;
  out[i] = (__bf16)eluf(buf[i] + bias[i & 511]);
}

// coef[t,e,m] = leaky( feat[edge_dst[t,e]] . Wc[t,m] + bc[t,m] )
__global__ void k_coef_rel(const float* __restrict__ feat, const int* __restrict__ edst,
                           const float* __restrict__ Wc, const float* __restrict__ bc,
                           float* __restrict__ coef) {
  int i = blockIdx.x * blockDim.x + threadIdx.x;
  if (i >= TT * ET * MM) return;
  int m = i & 7, te = i >> 3, t = te / ET;
  const float* hd = feat + (size_t)edst[te] * 128;
  const float* wc = Wc + (size_t)(t * 8 + m) * 128;
  float s = bc[t * 8 + m];
  for (int d = 0; d < 128; ++d) s += hd[d] * wc[d];
  coef[i] = leakyf(s);
}

// gather + convert: HS(bf16)[e,:] = feat[edge_src[e], :]
__global__ void k_gather_bf(const float* __restrict__ feat, const int* __restrict__ idx,
                            __bf16* __restrict__ out, int n) {
  int i = blockIdx.x * blockDim.x + threadIdx.x;
  if (i >= n * 16) return;                     // 8 floats per thread
  int e = i >> 4, q = i & 15;
  const float4* p = reinterpret_cast<const float4*>(feat + (size_t)idx[e] * 128) + q * 2;
  float4 a = p[0], b = p[1];
  v8bf o;
  o[0]=(__bf16)a.x; o[1]=(__bf16)a.y; o[2]=(__bf16)a.z; o[3]=(__bf16)a.w;
  o[4]=(__bf16)b.x; o[5]=(__bf16)b.y; o[6]=(__bf16)b.z; o[7]=(__bf16)b.w;
  reinterpret_cast<v8bf*>(out)[(size_t)e * 16 + q] = o;
}

__global__ void k_scatter(const float* __restrict__ msg, const int* __restrict__ didx,
                          float* __restrict__ sb, float* __restrict__ cnt, int n) {
  int w = (int)((blockIdx.x * blockDim.x + threadIdx.x) >> 5);
  int lane = threadIdx.x & 31;
  if (w >= n) return;
  int d = didx[w];
  const float* m = msg + (size_t)w * 128;
  float* o = sb + (size_t)d * 128;
  for (int j = lane; j < 128; j += 32) atomicAdd(&o[j], m[j]);
  if (lane == 0) atomicAdd(&cnt[d], 1.0f);
}

// rep[n] = layernorm(s[n]/max(cnt,1)) * ln_w + ln_b + h_bias   (wave per node)
__global__ void k_mem_rep(const float* __restrict__ s, const float* __restrict__ cnt,
                          const float* __restrict__ lnw, const float* __restrict__ lnb,
                          const float* __restrict__ hb, float* __restrict__ rep, int n) {
  int w = (int)((blockIdx.x * blockDim.x + threadIdx.x) >> 5);
  int lane = threadIdx.x & 31;
  if (w >= n) return;
  float c = fmaxf(cnt[w], 1.0f);
  const float* row = s + (size_t)w * 128;
  float x[4], sum = 0.f;
#pragma unroll
  for (int j = 0; j < 4; ++j) { x[j] = row[lane + 32 * j] / c; sum += x[j]; }
  for (int o = 16; o; o >>= 1) sum += __shfl_xor(sum, o);
  float mu = sum * (1.f / 128.f), var = 0.f;
#pragma unroll
  for (int j = 0; j < 4; ++j) { float d = x[j] - mu; var += d * d; }
  for (int o = 16; o; o >>= 1) var += __shfl_xor(var, o);
  float inv = rsqrtf(var * (1.f / 128.f) + 1e-5f);
#pragma unroll
  for (int j = 0; j < 4; ++j) {
    int d = lane + 32 * j;
    rep[(size_t)w * 128 + d] = (x[j] - mu) * inv * lnw[d] + lnb[d] + hb[d];
  }
}

__global__ void k_ncoef(const float* __restrict__ feat, const float* __restrict__ Wc,
                        const float* __restrict__ bc, float* __restrict__ out, int n) {
  int i = blockIdx.x * blockDim.x + threadIdx.x;
  if (i >= n * 8) return;
  int nn = i >> 3, m = i & 7;
  const float* f = feat + (size_t)nn * 128;
  const float* w = Wc + (size_t)m * 128;
  float s = bc[m];
  for (int d = 0; d < 128; ++d) s += f[d] * w[d];
  out[i] = leakyf(s);
}

__global__ void k_finish(const float* __restrict__ rep, const float* __restrict__ trans,
                         float* __restrict__ fnext, int total) {
  int i = blockIdx.x * blockDim.x + threadIdx.x;
  if (i >= total) return;
  fnext[i] = leakyf(rep[i]) + trans[i];
}

// final layernorm over concat of the 3 N x 128 blocks (wave per node)
__global__ void k_final_ln(const float* __restrict__ allE, const float* __restrict__ w,
                           const float* __restrict__ b, float* __restrict__ out, int n) {
  int wv = (int)((blockIdx.x * blockDim.x + threadIdx.x) >> 5);
  int lane = threadIdx.x & 31;
  if (wv >= n) return;
  float x[12], sum = 0.f;
#pragma unroll
  for (int k = 0; k < 12; ++k) {
    int j = lane + 32 * k;
    x[k] = allE[(size_t)(j >> 7) * NN * 128 + (size_t)wv * 128 + (j & 127)];
    sum += x[k];
  }
  for (int o = 16; o; o >>= 1) sum += __shfl_xor(sum, o);
  float mu = sum * (1.f / 384.f), var = 0.f;
#pragma unroll
  for (int k = 0; k < 12; ++k) { float d = x[k] - mu; var += d * d; }
  for (int o = 16; o; o >>= 1) var += __shfl_xor(var, o);
  float inv = rsqrtf(var * (1.f / 384.f) + 1e-5f);
#pragma unroll
  for (int k = 0; k < 12; ++k) {
    int j = lane + 32 * k;
    out[(size_t)wv * 384 + j] = (x[k] - mu) * inv * w[j] + b[j];
  }
}

// ---------------------------------------------------------------------------
static inline int nb(long long total) { return (int)((total + 255) / 256); }

static inline void gemm(const __bf16* A, const __bf16* B, float* C, int Mr, int Nc, int K,
                        const float* bias, int act, const float* rowscale, int rsStride,
                        int accumulate, hipStream_t s) {
  long long waves = (long long)(Mr / 16) * (Nc / 64);   // 16x64 tile per wave
  k_gemm<<<nb(waves * 32), 256, 0, s>>>(A, B, C, Mr, Nc, K, bias, act,
                                        rowscale, rsStride, accumulate);
}

extern "C" void kernel_launch(void* const* d_in, const int* in_sizes, int n_in,
                              void* d_out, int out_size, void* d_ws, size_t ws_size,
                              hipStream_t stream) {
  (void)in_sizes; (void)n_in; (void)out_size; (void)ws_size;
  const float* embedding   = (const float*)d_in[0];
  const float* gat_fc_W    = (const float*)d_in[1];
  const float* gat_attn_l  = (const float*)d_in[2];
  const float* gat_attn_r  = (const float*)d_in[3];
  const float* gat_bias    = (const float*)d_in[4];
  const float* proj_W      = (const float*)d_in[5];
  const float* proj_b      = (const float*)d_in[6];
  const float* mem_rel_Wc  = (const float*)d_in[7];
  const float* mem_rel_bc  = (const float*)d_in[8];
  const float* mem_rel_Ww  = (const float*)d_in[9];
  const float* mem_node_Wc = (const float*)d_in[10];
  const float* mem_node_bc = (const float*)d_in[11];
  const float* mem_node_Ww = (const float*)d_in[12];
  const float* mem_h_bias  = (const float*)d_in[13];
  const float* mem_ln_w    = (const float*)d_in[14];
  const float* mem_ln_b    = (const float*)d_in[15];
  const float* final_ln_w  = (const float*)d_in[16];
  const float* final_ln_b  = (const float*)d_in[17];
  const int*   src_u       = (const int*)d_in[18];
  const int*   dst_i       = (const int*)d_in[19];
  const int*   edge_src    = (const int*)d_in[20];
  const int*   edge_dst    = (const int*)d_in[21];
  float* out = (float*)d_out;

  // ---- workspace layout: f32 region then bf16 region (16B aligned) ----
  float* base = (float*)d_ws;
  size_t off = 0;
  auto take = [&](size_t nelem) -> float* {
    float* p = base + off; off += (nelem + 3) & ~(size_t)3; return p;
  };
  float*    allE   = take((size_t)3 * NN * DD);        // all_emb[0..2]
  float*    fsb    = take((size_t)20000 * HD);         // fs
  float*    fdb    = take((size_t)20000 * HD);         // fd, reused as rst
  float*    elb    = take(80000);
  float*    erb    = take(80000);
  float*    esc    = take(160000);
  float*    exb    = take(160000);
  unsigned* emk    = (unsigned*)take(80000);
  float*    den    = take(80000);
  float*    trans  = take((size_t)NN * DD);
  float*    HSf    = take(4);                          // (alignment pad)
  float*    MSGb   = take((size_t)ET * DD);
  float*    coefb  = take((size_t)TT * ET * MM);
  float*    sb     = take((size_t)NN * DD);
  float*    cntb   = take(NN);
  float*    repb   = take((size_t)NN * DD);
  float*    ncoefb = take((size_t)NN * MM);
  (void)HSf;

  __bf16* bbase = (__bf16*)(base + off);
  size_t boff = 0;
  auto takeb = [&](size_t nelem) -> __bf16* {
    __bf16* p = bbase + boff; boff += (nelem + 7) & ~(size_t)7; return p;
  };
  constexpr size_t NFC  = (size_t)2 * 2 * HD * DD;     // gat_fc_W
  constexpr size_t NPJ  = (size_t)2 * 2 * DD * HD;     // proj_W
  constexpr size_t NRW  = (size_t)2 * TT * MM * DD * DD; // mem_rel_Ww
  constexpr size_t NNW  = (size_t)2 * MM * DD * DD;    // mem_node_Ww
  __bf16* wFc   = takeb(NFC);
  __bf16* wPj   = takeb(NPJ);
  __bf16* wRel  = takeb(NRW);
  __bf16* wNod  = takeb(NNW);
  __bf16* featB = takeb((size_t)NN * DD);              // feat in bf16 (per layer)
  __bf16* guB   = takeb((size_t)20000 * HD);           // elu(rst+bias) in bf16
  __bf16* HSB   = takeb((size_t)ET * DD);              // gathered hs in bf16

  // ---- one-time weight conversions ----
  k_cvt<<<nb(NFC / 8), 256, 0, stream>>>(gat_fc_W, wFc, (int)(NFC / 8));
  k_cvt<<<nb(NPJ / 8), 256, 0, stream>>>(proj_W, wPj, (int)(NPJ / 8));
  k_cvt<<<nb(NRW / 8), 256, 0, stream>>>(mem_rel_Ww, wRel, (int)(NRW / 8));
  k_cvt<<<nb(NNW / 8), 256, 0, stream>>>(mem_node_Ww, wNod, (int)(NNW / 8));

  hipMemcpyAsync(allE, embedding, (size_t)NN * DD * sizeof(float),
                 hipMemcpyDeviceToDevice, stream);

  for (int l = 0; l < 2; ++l) {
    const float* feat  = allE + (size_t)l * NN * DD;
    float*       featN = allE + (size_t)(l + 1) * NN * DD;

    k_cvt<<<nb((long long)NN * DD / 8), 256, 0, stream>>>(feat, featB, NN * DD / 8);

    // ---- two GAT sides: s=0 (items->users), s=1 (users->items) ----
    for (int s = 0; s < 2; ++s) {
      const __bf16* hsrcB = featB + (size_t)(s == 0 ? NU : 0) * DD;
      const __bf16* hdstB = featB + (size_t)(s == 0 ? 0 : NU) * DD;
      int nsrc = (s == 0) ? NI : NU;
      int ndst = (s == 0) ? NU : NI;
      const int* sidx = (s == 0) ? dst_i : src_u;
      const int* didx = (s == 0) ? src_u : dst_i;
      const __bf16* W  = wFc + (size_t)(l * 2 + s) * HD * DD;
      const float* al = gat_attn_l + (size_t)(l * 2 + s) * 4 * DD;
      const float* ar = gat_attn_r + (size_t)(l * 2 + s) * 4 * DD;
      const float* gb = gat_bias   + (size_t)(l * 2 + s) * HD;
      const __bf16* pW = wPj + (size_t)(l * 2 + s) * DD * HD;
      const float* pb = proj_b     + (size_t)(l * 2 + s) * DD;
      float* tdst = trans + (size_t)(s == 0 ? 0 : NU) * DD;

      gemm(hsrcB, W, fsb, nsrc, HD, DD, nullptr, 0, nullptr, 0, 0, stream);  // fs
      gemm(hdstB, W, fdb, ndst, HD, DD, nullptr, 0, nullptr, 0, 0, stream);  // fd
      k_attn_score<<<nb(nsrc * 4), 256, 0, stream>>>(fsb, al, elb, nsrc * 4);
      k_attn_score<<<nb(ndst * 4), 256, 0, stream>>>(fdb, ar, erb, ndst * 4);
      hipMemsetAsync(emk, 0, (size_t)ndst * 4 * sizeof(unsigned), stream);   // key(-inf)>0
      hipMemsetAsync(den, 0, (size_t)ndst * 4 * sizeof(float), stream);
      k_edge_max<<<nb((long long)EUI * 4), 256, 0, stream>>>(elb, erb, sidx, didx, esc, emk, EUI);
      k_edge_exp<<<nb((long long)EUI * 4), 256, 0, stream>>>(esc, emk, didx, exb, den, EUI);
      hipMemsetAsync(fdb, 0, (size_t)ndst * HD * sizeof(float), stream);     // fd -> rst
      k_gat_agg<<<nb((long long)EUI * 4 * 32), 256, 0, stream>>>(exb, den, fsb, sidx, didx, fdb, EUI);
      k_bias_elu_bf<<<nb((long long)ndst * HD), 256, 0, stream>>>(fdb, gb, guB, ndst * HD);
      gemm(guB, pW, tdst, ndst, DD, HD, pb, 1, nullptr, 0, 0, stream);       // ru / ri
    }

    // ---- memory layer ----
    k_coef_rel<<<nb((long long)TT * ET * MM), 256, 0, stream>>>(
        feat, edge_dst, mem_rel_Wc + (size_t)l * TT * MM * DD,
        mem_rel_bc + (size_t)l * TT * MM, coefb);
    hipMemsetAsync(sb, 0, (size_t)NN * DD * sizeof(float), stream);
    hipMemsetAsync(cntb, 0, (size_t)NN * sizeof(float), stream);
    for (int t = 0; t < TT; ++t) {
      k_gather_bf<<<nb((long long)ET * 16), 256, 0, stream>>>(feat, edge_src + (size_t)t * ET, HSB, ET);
      for (int m = 0; m < MM; ++m) {
        const __bf16* Wm = wRel + (((size_t)l * TT + t) * MM + m) * DD * DD;
        // MSG (+)= coef[t,:,m] * (HS @ Wm^T)
        gemm(HSB, Wm, MSGb, ET, DD, DD, nullptr, 0,
             coefb + (size_t)t * ET * MM + m, MM, m > 0, stream);
      }
      k_scatter<<<nb((long long)ET * 32), 256, 0, stream>>>(MSGb, edge_dst + (size_t)t * ET, sb, cntb, ET);
    }
    k_mem_rep<<<nb((long long)NN * 32), 256, 0, stream>>>(
        sb, cntb, mem_ln_w + (size_t)l * DD, mem_ln_b + (size_t)l * DD,
        mem_h_bias + (size_t)l * DD, repb, NN);
    k_ncoef<<<nb((long long)NN * MM), 256, 0, stream>>>(
        feat, mem_node_Wc + (size_t)l * MM * DD, mem_node_bc + (size_t)l * MM, ncoefb, NN);
    for (int m = 0; m < MM; ++m) {
      const __bf16* Wm = wNod + ((size_t)l * MM + m) * DD * DD;
      gemm(featB, Wm, repb, NN, DD, DD, nullptr, 0, ncoefb + m, MM, 1, stream);  // rep += ...
    }
    k_finish<<<nb((long long)NN * DD), 256, 0, stream>>>(repb, trans, featN, NN * DD);
  }

  k_final_ln<<<nb((long long)NN * 32), 256, 0, stream>>>(allE, final_ln_w, final_ln_b, out, NN);
}
